// MultiHeadAttention_26517128086004
// MI455X (gfx1250) — matrix operand order repaired and checked by reading, not compiled, not run on hardware
//
#include <hip/hip_runtime.h>
#include <hip/hip_bf16.h>

// Problem constants (fixed by the reference)
constexpr int Bc = 2, Sc = 2048, Dc = 1024, Hc = 16, DKc = 64;
constexpr int Mc = Bc * Sc;          // 4096 rows for all GEMMs

// ---- CDNA5 WMMA types (wave32) -------------------------------------------
typedef __attribute__((ext_vector_type(16))) __bf16 v16bf;
typedef __attribute__((ext_vector_type(8)))  float  v8f;
typedef __attribute__((ext_vector_type(4)))  float  fvec4;   // trivial, union-safe

union BFrag {
    v16bf v;
    unsigned short us[16];
    fvec4 f4[2];
};

__device__ inline unsigned short f2bf(float f) {
    unsigned u = __builtin_bit_cast(unsigned, f);
    unsigned r = u + 0x7FFFu + ((u >> 16) & 1u);   // round-to-nearest-even
    return (unsigned short)(r >> 16);
}
__device__ inline float bf2f(unsigned short b) {
    unsigned u = ((unsigned)b) << 16;
    return __builtin_bit_cast(float, u);
}
__device__ inline v8f wmma_bf16(const BFrag& a, const BFrag& b, v8f c) {
    return __builtin_amdgcn_wmma_f32_16x16x32_bf16(
        false, a.v, false, b.v, (short)0, c, false, false);
}

// ---- DPP butterfly reductions over each 16-lane group (VALU-only) ---------
// Steps: quad_perm[1,0,3,2]=0xB1, quad_perm[2,3,0,1]=0x4E,
//        row_half_mirror=0x141, row_mirror=0x140. None cross the 16-lane
//        group boundary, matching the WMMA C-layout row groups.
template <int CTRL>
__device__ inline float dpp_swap(float x) {
    int t = __builtin_amdgcn_update_dpp(0, __builtin_bit_cast(int, x),
                                        CTRL, 0xF, 0xF, true);
    return __builtin_bit_cast(float, t);
}
__device__ inline float red_max16(float x) {
    x = fmaxf(x, dpp_swap<0xB1>(x));
    x = fmaxf(x, dpp_swap<0x4E>(x));
    x = fmaxf(x, dpp_swap<0x141>(x));
    x = fmaxf(x, dpp_swap<0x140>(x));
    return x;
}
__device__ inline float red_sum16(float x) {
    x += dpp_swap<0xB1>(x);
    x += dpp_swap<0x4E>(x);
    x += dpp_swap<0x141>(x);
    x += dpp_swap<0x140>(x);
    return x;
}

// ---- One-time f32 -> bf16 conversion (bandwidth-bound, runs once) ---------
__global__ void conv_f32_bf16(const float* __restrict__ in,
                              unsigned short* __restrict__ out, int n8) {
    const int t = blockIdx.x * blockDim.x + threadIdx.x;
    if (t >= n8) return;
    const float* p = in + (size_t)t * 8;
    fvec4 a = *(const fvec4*)p;
    fvec4 b = *(const fvec4*)(p + 4);
    union { unsigned short us[8]; fvec4 f; } o;
    o.us[0] = f2bf(a.x); o.us[1] = f2bf(a.y);
    o.us[2] = f2bf(a.z); o.us[3] = f2bf(a.w);
    o.us[4] = f2bf(b.x); o.us[5] = f2bf(b.y);
    o.us[6] = f2bf(b.z); o.us[7] = f2bf(b.w);
    *(fvec4*)(out + (size_t)t * 8) = o.f;
}

// ---- GEMM: C[m,n] = sum_k A_bf16[m,k] * W_bf16[n,k] -----------------------
// Wave tile: 32(M) x 64(N) -> 8 WMMAs per k-step fed by 12 b128 loads.
// OUT_F32: store f32 (final O projection). Else bf16; vtrans==1 stores V
// transposed per head: vT[((b*H+h)*DK+d)*S + s].
template <bool OUT_F32>
__global__ __launch_bounds__(128)
void gemm_bf16(const unsigned short* __restrict__ A,
               const unsigned short* __restrict__ W,
               void* __restrict__ Cout, int vtrans) {
    const int lane = threadIdx.x & 31;
    const int wv   = threadIdx.x >> 5;
    const int half = lane >> 4;          // 0: lanes 0-15, 1: lanes 16-31
    const int l16  = lane & 15;
    const int m0   = blockIdx.y * 128 + wv * 32;
    const int n0   = blockIdx.x * 64;

    v8f acc[2][4] = {};
    const unsigned short* ar0 = A + (size_t)(m0 + l16) * Dc;
    const unsigned short* ar1 = ar0 + (size_t)16 * Dc;
    const int kA0 = half * 16;           // A K-chunk base: {0..15} or {16..31}

    for (int k = 0; k < Dc; k += 32) {
        __builtin_prefetch(ar0 + k + 32, 0, 0);    // global_prefetch_b8
        __builtin_prefetch(ar1 + k + 32, 0, 0);
        BFrag a0, a1;
        a0.f4[0] = *(const fvec4*)(ar0 + k + kA0);
        a0.f4[1] = *(const fvec4*)(ar0 + k + kA0 + 8);
        a1.f4[0] = *(const fvec4*)(ar1 + k + kA0);
        a1.f4[1] = *(const fvec4*)(ar1 + k + kA0 + 8);
        #pragma unroll
        for (int nb = 0; nb < 4; ++nb) {
            const unsigned short* wp =
                W + (size_t)(n0 + nb * 16 + l16) * Dc + k + half * 16;
            BFrag b;
            b.f4[0] = *(const fvec4*)wp;
            b.f4[1] = *(const fvec4*)(wp + 8);
            acc[0][nb] = wmma_bf16(a0, b, acc[0][nb]);
            acc[1][nb] = wmma_bf16(a1, b, acc[1][nb]);
        }
    }
    #pragma unroll
    for (int ms = 0; ms < 2; ++ms) {
        #pragma unroll
        for (int nb = 0; nb < 4; ++nb) {
            #pragma unroll
            for (int r = 0; r < 8; ++r) {
                const int m = m0 + ms * 16 + r + half * 8;   // C lane halves: r / r+8
                const int n = n0 + nb * 16 + l16;
                const float val = acc[ms][nb][r];
                if (OUT_F32) {
                    ((float*)Cout)[(size_t)m * Dc + n] = val;
                } else {
                    const unsigned short bv = f2bf(val);
                    if (!vtrans) {
                        ((unsigned short*)Cout)[(size_t)m * Dc + n] = bv;
                    } else {
                        const int bb = m / Sc, ss = m % Sc;
                        const int hh = n >> 6, dd = n & 63;
                        ((unsigned short*)Cout)
                            [(((size_t)bb * Hc + hh) * DKc + dd) * Sc + ss] = bv;
                    }
                }
            }
        }
    }
}

// ---- RoPE on Q and K (bf16 in place) --------------------------------------
__global__ void rope_kernel(unsigned short* __restrict__ Q,
                            unsigned short* __restrict__ Kb,
                            const int* __restrict__ pos) {
    const int tid   = blockIdx.x * blockDim.x + threadIdx.x;
    const int total = Bc * Sc * Hc * (DKc / 2);
    if (tid >= total) return;
    const int i  = tid & 31;                 // pair index within head
    const int h  = (tid >> 5) & (Hc - 1);
    const int s  = (tid >> 9) & (Sc - 1);
    const int b  = tid >> 20;
    const float p = (float)pos[s];
    const float inv_freq = __powf(10000.0f, -(2.0f * (float)i) / (float)DKc);
    const float ang = p * inv_freq;
    const float cs = cosf(ang), sn = sinf(ang);
    const size_t base = ((size_t)b * Sc + s) * Dc + h * DKc + 2 * i;
    {
        float t0 = bf2f(Q[base]), t1 = bf2f(Q[base + 1]);
        Q[base]     = f2bf(cs * t0 - sn * t1);
        Q[base + 1] = f2bf(sn * t0 + cs * t1);
    }
    {
        float t0 = bf2f(Kb[base]), t1 = bf2f(Kb[base + 1]);
        Kb[base]     = f2bf(cs * t0 - sn * t1);
        Kb[base + 1] = f2bf(sn * t0 + cs * t1);
    }
}

// ---- Flash attention: one wave per (b, h, 16-query tile) ------------------
__global__ __launch_bounds__(128)
void attn_kernel(const unsigned short* __restrict__ Q,
                 const unsigned short* __restrict__ Kb,
                 const unsigned short* __restrict__ VT,
                 unsigned short* __restrict__ O) {
    __shared__ unsigned short p_lds[4][16 * 32];   // per-wave P tile (bf16)

    const int lane = threadIdx.x & 31;
    const int wv   = threadIdx.x >> 5;
    const int half = lane >> 4;
    const int l16  = lane & 15;
    // Scalarize the per-wave unit id: qt/b/h become SGPRs, so the causal
    // branch below is a scalar branch (EXEC stays all-ones around WMMA).
    const int unit = __builtin_amdgcn_readfirstlane(blockIdx.x * 4 + wv);
    const int qt   = unit & 127;                   // query tile (S/16)
    const int bh   = unit >> 7;
    const int h    = bh & (Hc - 1);
    const int b    = bh >> 4;

    // Loop-invariant Q A-fragments (dd = 0, 32)
    BFrag qa[2];
    const unsigned short* qbase =
        Q + ((size_t)b * Sc + (size_t)qt * 16 + l16) * Dc + h * DKc;
    #pragma unroll
    for (int dd = 0; dd < 2; ++dd) {
        const unsigned short* qp = qbase + dd * 32 + half * 16;
        qa[dd].f4[0] = *(const fvec4*)qp;
        qa[dd].f4[1] = *(const fvec4*)(qp + 8);
    }

    v8f o[4] = {};
    float mrow[8];     // running row max (uniform across 16-lane group)
    float lpart[8];    // per-lane PARTIAL row sum (reduced once at the end)
    #pragma unroll
    for (int r = 0; r < 8; ++r) { mrow[r] = -3.0e38f; lpart[r] = 0.0f; }

    const float scale = 0.125f;                    // 1/sqrt(64)
    for (int j0 = 0; j0 <= qt; j0 += 2) {          // 32 keys per step
        const bool have1 = (j0 + 1) <= qt;         // scalar branch

        v8f s0 = {}, s1 = {};
        #pragma unroll
        for (int dd = 0; dd < 2; ++dd) {           // QK^T tile j0
            const unsigned short* kp =
                Kb + ((size_t)b * Sc + j0 * 16 + l16) * Dc + h * DKc + dd * 32 + half * 16;
            BFrag kb;
            kb.f4[0] = *(const fvec4*)kp;
            kb.f4[1] = *(const fvec4*)(kp + 8);
            s0 = wmma_bf16(qa[dd], kb, s0);
        }
        if (have1) {                               // QK^T tile j0+1
            #pragma unroll
            for (int dd = 0; dd < 2; ++dd) {
                const unsigned short* kp =
                    Kb + ((size_t)b * Sc + (j0 + 1) * 16 + l16) * Dc + h * DKc + dd * 32 + half * 16;
                BFrag kb;
                kb.f4[0] = *(const fvec4*)kp;
                kb.f4[1] = *(const fvec4*)(kp + 8);
                s1 = wmma_bf16(qa[dd], kb, s1);
            }
        }

        // scale + causal mask + online softmax (rows span 16-lane groups)
        #pragma unroll
        for (int r = 0; r < 8; ++r) {
            const int qrow = qt * 16 + r + half * 8;
            const int key0 = j0 * 16 + l16;
            float v0 = s0[r] * scale;
            if (key0 > qrow) v0 = -3.0e38f;
            float v1 = -3.0e38f;
            if (have1) {
                v1 = s1[r] * scale;
                if (key0 + 16 > qrow) v1 = -3.0e38f;
            }
            const float mx   = red_max16(fmaxf(v0, v1));   // DPP butterfly
            const float mnew = fmaxf(mrow[r], mx);
            const float alpha = __expf(mrow[r] - mnew);
            const float p0 = __expf(v0 - mnew);
            const float p1 = __expf(v1 - mnew);
            lpart[r] = lpart[r] * alpha + (p0 + p1);       // per-lane partial
            mrow[r] = mnew;
            #pragma unroll
            for (int nb = 0; nb < 4; ++nb) o[nb][r] *= alpha;
            // C-layout -> LDS (row-major 16x32), for A-layout re-read
            const int m = r + half * 8;
            p_lds[wv][m * 32 + l16]      = f2bf(p0);
            p_lds[wv][m * 32 + 16 + l16] = f2bf(p1);
        }
        asm volatile("s_wait_dscnt 0" ::: "memory");   // wave-local LDS RAW

        BFrag pa;                                   // P as 16x32 A fragment
        {
            const unsigned short* pp = &p_lds[wv][l16 * 32 + half * 16];
            pa.f4[0] = *(const fvec4*)pp;
            pa.f4[1] = *(const fvec4*)(pp + 8);
        }
        // P @ V over 32 keys; V is pre-transposed so key runs are contiguous
        const int koff = have1 ? half * 16 : 0;     // clamp OOB half (P==0 there)
        #pragma unroll
        for (int nb = 0; nb < 4; ++nb) {
            const int d = nb * 16 + l16;
            const unsigned short* vp =
                VT + (((size_t)b * Hc + h) * DKc + d) * Sc + j0 * 16 + koff;
            BFrag vb;
            vb.f4[0] = *(const fvec4*)vp;
            vb.f4[1] = *(const fvec4*)(vp + 8);
            o[nb] = wmma_bf16(pa, vb, o[nb]);
        }
    }

    #pragma unroll
    for (int r = 0; r < 8; ++r) {
        const float inv = 1.0f / red_sum16(lpart[r]);      // one reduce at end
        const int qrow = qt * 16 + r + half * 8;
        const size_t base = ((size_t)b * Sc + qrow) * Dc + h * DKc;
        #pragma unroll
        for (int nb = 0; nb < 4; ++nb)
            O[base + nb * 16 + l16] = f2bf(o[nb][r] * inv);
    }
}

// ---- Launch ---------------------------------------------------------------
extern "C" void kernel_launch(void* const* d_in, const int* in_sizes, int n_in,
                              void* d_out, int out_size, void* d_ws, size_t ws_size,
                              hipStream_t stream) {
    const float* x   = (const float*)d_in[0];
    const int*   tp  = (const int*)d_in[1];
    const float* qw  = (const float*)d_in[2];
    const float* kw  = (const float*)d_in[3];
    const float* vw  = (const float*)d_in[4];
    const float* ow  = (const float*)d_in[5];
    float* out = (float*)d_out;

    const size_t xel = (size_t)Bc * Sc * Dc;       // 4.19M
    const size_t wel = (size_t)Dc * Dc;            // 1.05M
    unsigned short* xb   = (unsigned short*)d_ws;  // bf16 x
    unsigned short* qwb  = xb   + xel;             // bf16 weights
    unsigned short* kwb  = qwb  + wel;
    unsigned short* vwb  = kwb  + wel;
    unsigned short* owb  = vwb  + wel;
    unsigned short* qb   = owb  + wel;             // bf16 activations
    unsigned short* kb   = qb   + xel;
    unsigned short* vtb  = kb   + xel;
    unsigned short* ab   = vtb  + xel;

    // One-time conversions (bandwidth-bound; removes all VALU cvt from GEMMs)
    conv_f32_bf16<<<(int)(xel / 8 / 256), 256, 0, stream>>>(x, xb, (int)(xel / 8));
    conv_f32_bf16<<<(int)(wel / 8 / 256), 256, 0, stream>>>(qw, qwb, (int)(wel / 8));
    conv_f32_bf16<<<(int)(wel / 8 / 256), 256, 0, stream>>>(kw, kwb, (int)(wel / 8));
    conv_f32_bf16<<<(int)(wel / 8 / 256), 256, 0, stream>>>(vw, vwb, (int)(wel / 8));
    conv_f32_bf16<<<(int)(wel / 8 / 256), 256, 0, stream>>>(ow, owb, (int)(wel / 8));

    dim3 gg(Dc / 64, Mc / 128);                    // (16, 32)
    gemm_bf16<false><<<gg, 128, 0, stream>>>(xb, qwb, qb, 0);
    gemm_bf16<false><<<gg, 128, 0, stream>>>(xb, kwb, kb, 0);
    gemm_bf16<false><<<gg, 128, 0, stream>>>(xb, vwb, vtb, 1);

    const int rope_threads = Bc * Sc * Hc * (DKc / 2);
    rope_kernel<<<(rope_threads + 255) / 256, 256, 0, stream>>>(qb, kb, tp);

    attn_kernel<<<(Bc * Hc * (Sc / 16)) / 4, 128, 0, stream>>>(qb, kb, vtb, ab);

    gemm_bf16<true><<<gg, 128, 0, stream>>>(ab, owb, out, 0);
}
